// EdgeLoss_52183852646534
// MI455X (gfx1250) — compile-verified, matrix operand order checked by reading, not run on hardware
//
#include <hip/hip_runtime.h>
#include <hip/hip_bf16.h>
#include <stdint.h>

// MI455X (gfx1250) notes:
//  - HBM-bound problem (~110 MB total traffic -> ~5us floor @ 23.3 TB/s).
//  - wave32; 256-thread blocks = 8 waves.
//  - LDS-staged fused Canny pipeline; bit-packed local hysteresis (exact:
//    10 iterations spread <= 10 px, halo = 10).
//  - Final reduction runs through v_wmma_f32_16x16x32_f16 (A = per-lane
//    popcounts, B = ones -> row sums), then shfl tree + atomicAdd.

typedef __attribute__((ext_vector_type(16))) _Float16 v16h;
typedef __attribute__((ext_vector_type(8)))  float    v8f;

#define IMH 512
#define IMW 512
#define TILE1 32
#define TILE2 64
#define RAD 10
#define WIN (TILE2 + 2 * RAD)   // 84 rows, 84 cols -> 3 u32 words per row

__device__ __forceinline__ int refl(int i, int n) {
    i = (i < 0) ? -i : i;
    return (i >= n) ? (2 * n - 2 - i) : i;
}

// ---------------------------------------------------------------------------
// Kernel 1: fused gray -> gaussian5 (reflect) -> sobel (reflect) -> mag/dir
//           -> NMS (zero pad) -> double threshold. Output uint8 {0,1,2}.
// One block computes a 32x32 tile; z indexes 2 images x 16 batch = 32 maps.
// ---------------------------------------------------------------------------
__global__ __launch_bounds__(256) void canny_thresh_kernel(
    const float* __restrict__ yhat, const float* __restrict__ yref,
    uint8_t* __restrict__ tmap)
{
    const int z = blockIdx.z;  // 0..31
    const float* img = (z < 16) ? (yhat + (size_t)z * 3 * IMH * IMW)
                                : (yref + (size_t)(z - 16) * 3 * IMH * IMW);
    uint8_t* out = tmap + (size_t)z * IMH * IMW;
    const int gy0 = blockIdx.y * TILE1;
    const int gx0 = blockIdx.x * TILE1;
    const int tid = threadIdx.x;

    __shared__ float   sg[40][41];   // gray, halo 4 (reflect-loaded)
    __shared__ float   shz[40][37];  // horizontal blur, 40 x 36
    __shared__ float   sb[36][37];   // full blur, 36 x 36
    __shared__ float   sm[34][35];   // gradient magnitude, halo 1 (0 outside image)
    __shared__ uint8_t sdir[34][35]; // direction bin 0..3

    // gaussian5 sigma=1 normalized weights (fp32, matches reference to ulp-ish)
    const float w0 = 0.40261994689424435f;
    const float w1 = 0.24420134159500501f;
    const float w2 = 0.05448868495787246f;

    // Stage A: gray with reflect indexing (coalesced along x)
    for (int t = tid; t < 40 * 40; t += 256) {
        int wy = t / 40, wx = t % 40;
        int ry = refl(gy0 - 4 + wy, IMH);
        int rx = refl(gx0 - 4 + wx, IMW);
        size_t base = (size_t)ry * IMW + rx;
        float r = img[base];
        float g = img[(size_t)IMH * IMW + base];
        float b = img[(size_t)2 * IMH * IMW + base];
        sg[wy][wx] = 0.299f * r + 0.587f * g + 0.114f * b;
    }
    __syncthreads();

    // Stage B: horizontal gaussian (reflect handled by Stage A layout)
    for (int t = tid; t < 40 * 36; t += 256) {
        int wy = t / 36, wx = t % 36;
        shz[wy][wx] = w2 * (sg[wy][wx]     + sg[wy][wx + 4])
                    + w1 * (sg[wy][wx + 1] + sg[wy][wx + 3])
                    + w0 *  sg[wy][wx + 2];
    }
    __syncthreads();

    // Stage C: vertical gaussian
    for (int t = tid; t < 36 * 36; t += 256) {
        int wy = t / 36, wx = t % 36;
        sb[wy][wx] = w2 * (shz[wy][wx]     + shz[wy + 4][wx])
                   + w1 * (shz[wy + 1][wx] + shz[wy + 3][wx])
                   + w0 *  shz[wy + 2][wx];
    }
    __syncthreads();

    // Stage D: sobel (cross-correlation) + magnitude + direction bin, halo 1.
    // Out-of-image cells get mag = 0 (zero-padded NMS neighborhood).
    for (int t = tid; t < 34 * 34; t += 256) {
        int my = t / 34, mx = t % 34;
        int gy = gy0 - 1 + my, gx = gx0 - 1 + mx;
        if ((unsigned)gy >= IMH || (unsigned)gx >= IMW) {
            sm[my][mx] = 0.0f; sdir[my][mx] = 0;
        } else {
            int by = my + 1, bx = mx + 1;   // blur-window coord of (gy,gx)
            float tl = sb[by - 1][bx - 1], tc = sb[by - 1][bx], tr = sb[by - 1][bx + 1];
            float ml = sb[by    ][bx - 1],                      mr = sb[by    ][bx + 1];
            float bl = sb[by + 1][bx - 1], bc = sb[by + 1][bx], br = sb[by + 1][bx + 1];
            float gxs = (tr - tl) + 2.0f * (mr - ml) + (br - bl);
            float gys = (bl - tl) + 2.0f * (bc - tc) + (br - tr);
            float mag = sqrtf(gxs * gxs + gys * gys + 1e-12f);
            float ang = atan2f(gys, gxs);
            int r4 = __float2int_rn(ang * 1.27323954473516268f); // ang / (pi/4), RNE
            sm[my][mx] = mag;
            sdir[my][mx] = (uint8_t)(((r4 % 4) + 4) & 3);
        }
    }
    __syncthreads();

    // Stage E: NMS + double threshold -> {0,1,2}
    const int dyp[4] = {0, 1, 1, 1};
    const int dxp[4] = {1, 1, 0, -1};
    for (int t = tid; t < TILE1 * TILE1; t += 256) {
        int ty = t / TILE1, tx = t % TILE1;
        int my = ty + 1, mx = tx + 1;
        float m = sm[my][mx];
        int b4 = sdir[my][mx];
        float n1 = sm[my + dyp[b4]][mx + dxp[b4]];
        float n2 = sm[my - dyp[b4]][mx - dxp[b4]];
        uint8_t code = 0;
        if (m >= n1 && m >= n2)
            code = (m >= 0.2f) ? 2 : ((m >= 0.1f) ? 1 : 0);
        out[(size_t)(gy0 + ty) * IMW + (gx0 + tx)] = code;
    }
}

// ---------------------------------------------------------------------------
// Kernel 2: local bit-packed hysteresis (10 iters, halo 10 -> exact) for both
// images of one batch entry, then XOR/popcount of strong masks over the 64x64
// center, WMMA row-sum reduction, atomicAdd of scaled partial into d_out.
// ---------------------------------------------------------------------------
__global__ __launch_bounds__(256) void hyst_loss_kernel(
    const uint8_t* __restrict__ tmap, float* __restrict__ loss)
{
    const int n  = blockIdx.z;                  // batch 0..15
    const int oy = blockIdx.y * TILE2 - RAD;    // window origin (global)
    const int ox = blockIdx.x * TILE2 - RAD;
    const int tid = threadIdx.x;
    const uint8_t* mapA = tmap + (size_t)n * IMH * IMW;
    const uint8_t* mapB = tmap + (size_t)(16 + n) * IMH * IMW;

    __shared__ uint32_t SA[WIN][3], WA[WIN][3];  // strong / weak bitboards
    __shared__ uint32_t SB[WIN][3], WB[WIN][3];

    // Load + pack (zero outside image == zero-pad for hysteresis dilation)
    if (tid < WIN * 3) {
        int r = tid / 3, w = tid % 3;
        int gy = oy + r;
        uint32_t sa = 0, wa = 0, sb2 = 0, wb2 = 0;
        if ((unsigned)gy < IMH) {
            int xbase = ox + w * 32;
            const uint8_t* rowA = mapA + (size_t)gy * IMW;
            const uint8_t* rowB = mapB + (size_t)gy * IMW;
            __builtin_prefetch(rowB + ((xbase < 0) ? 0 : xbase), 0, 0); // global_prefetch_b8
            for (int k = 0; k < 32; ++k) {
                int gx = xbase + k;
                if ((unsigned)gx < IMW) {
                    uint32_t va = rowA[gx], vb = rowB[gx];
                    sa  |= (uint32_t)(va == 2) << k;
                    wa  |= (uint32_t)(va == 1) << k;
                    sb2 |= (uint32_t)(vb == 2) << k;
                    wb2 |= (uint32_t)(vb == 1) << k;
                }
            }
        }
        SA[r][w] = sa;  WA[r][w] = wa;
        SB[r][w] = sb2; WB[r][w] = wb2;
    }
    __syncthreads();

    // Synchronous (Jacobi) promotion: S |= W & dilate3x3(S), 10 iterations.
    auto run10 = [&](uint32_t (*S)[3], const uint32_t (*Wk)[3]) {
        const int r = tid / 3, w = tid % 3;
        for (int it = 0; it < 10; ++it) {
            uint32_t d = 0;
            if (tid < WIN * 3) {
                for (int rr = r - 1; rr <= r + 1; ++rr) {
                    if ((unsigned)rr < (unsigned)WIN) {
                        uint32_t c = S[rr][w];
                        uint32_t h = c | (c << 1) | (c >> 1);
                        if (w > 0) h |= S[rr][w - 1] >> 31;
                        if (w < 2) h |= S[rr][w + 1] << 31;
                        d |= h;
                    }
                }
            }
            __syncthreads();                       // all reads of old S done
            if (tid < WIN * 3) S[r][w] |= Wk[r][w] & d;
            __syncthreads();                       // all writes done
        }
    };
    run10(SA, WA);
    run10(SB, WB);

    // Diff count over the 64x64 center (window rows/cols [10,74)).
    const uint32_t cmask[3] = {0xFFFFFC00u, 0xFFFFFFFFu, 0x000003FFu};
    float cnt = 0.0f;
    if (tid < 192) {                // 64 rows x 3 words
        int r = RAD + tid / 3, w = tid % 3;
        cnt = (float)__popc((SA[r][w] ^ SB[r][w]) & cmask[w]);  // <= 32, f16-exact
    }

    // WMMA reduction: D = A x ones + 0 -> every D element sums a row of A,
    // so sum(D) = 16 * sum(A). Executed uniformly (EXEC all ones).
    v16h a = {};
    a[0] = (_Float16)cnt;
    v16h onesv;
#pragma unroll
    for (int i = 0; i < 16; ++i) onesv[i] = (_Float16)1.0f;
    v8f c = {};
    c = __builtin_amdgcn_wmma_f32_16x16x32_f16(
        /*neg_a=*/false, a, /*neg_b=*/false, onesv,
        /*c_mod=*/(short)0, c, /*reuse_a=*/false, /*reuse_b=*/false);
    float s = c[0] + c[1] + c[2] + c[3] + c[4] + c[5] + c[6] + c[7];
    for (int off = 16; off > 0; off >>= 1)
        s += __shfl_xor(s, off, 32);
    if ((tid & 31) == 0) {
        // s = 16 * wave_total; mean over 16*1*512*512 elements
        atomicAdd(loss, s * (1.0f / 16.0f) * (1.0f / 4194304.0f));
    }
}

// ---------------------------------------------------------------------------
extern "C" void kernel_launch(void* const* d_in, const int* in_sizes, int n_in,
                              void* d_out, int out_size, void* d_ws, size_t ws_size,
                              hipStream_t stream) {
    (void)in_sizes; (void)n_in; (void)out_size; (void)ws_size;
    const float* yhat = (const float*)d_in[0];
    const float* yref = (const float*)d_in[1];
    uint8_t* tmap = (uint8_t*)d_ws;              // 32 * 512 * 512 = 8 MB scratch
    float* loss = (float*)d_out;

    hipMemsetAsync(loss, 0, sizeof(float), stream);  // graph-capture safe

    dim3 g1(IMW / TILE1, IMH / TILE1, 32);           // 16 x 16 x 32 blocks
    canny_thresh_kernel<<<g1, 256, 0, stream>>>(yhat, yref, tmap);

    dim3 g2(IMW / TILE2, IMH / TILE2, 16);           // 8 x 8 x 16 blocks
    hyst_loss_kernel<<<g2, 256, 0, stream>>>(tmap, loss);
}
